// MultimodalGraphFusionNetwork_77996606096091
// MI455X (gfx1250) — matrix operand order repaired and checked by reading, not compiled, not run on hardware
//
#include <hip/hip_runtime.h>
#include <hip/hip_bf16.h>

// ---------------------------------------------------------------------------
// MultimodalGraphFusionNetwork for MI455X (gfx1250, wave32, WMMA + TDM).
//
//  * All GEMMs use v_wmma_f32_16x16x32_f16 (f16 in, f32 accumulate).
//  * GAT attention is fused flash-style (online softmax, accumulator rescale,
//    2 WMMAs per 32-wide j-chunk). The (B,N,N,H) score tensor is never
//    materialized; dominant HBM traffic is the adjacency (~450 MB total
//    ~= 20us at 23.3 TB/s).
//  * Tile staging uses the Tensor Data Mover (tensor_load_to_lds) with
//    double buffering (s_wait_tensorcnt pipelining). TDM OOB-returns-zero
//    handles the K tails (K=70/148) with no predication, and TDM LDS row
//    padding (pad_interval/pad_amount) breaks LDS bank conflicts on the
//    WMMA fragment gathers.
// ---------------------------------------------------------------------------

constexpr int kB     = 4;
constexpr int kL     = 384;
constexpr int kN     = 1152;   // 3*L
constexpr int kH     = 128;
constexpr int kHEADS = 4;
constexpr int kDIM   = 32;
constexpr int kDT    = 768;
constexpr int kDV2   = 70;     // 2*DV
constexpr int kDA2   = 148;    // 2*DA
constexpr float kNEG   = -1000000000.0f;
constexpr float kALPHA = 0.2f;

typedef __attribute__((ext_vector_type(16))) _Float16 v16h;
typedef __attribute__((ext_vector_type(8)))  float    v8f;

// ---------------------------------------------------------------------------
// Tensor Data Mover helpers
// ---------------------------------------------------------------------------
#if defined(__has_builtin)
#if __has_builtin(__builtin_amdgcn_tensor_load_to_lds) && \
    __has_builtin(__builtin_amdgcn_s_wait_tensorcnt)
#define USE_TDM 1
#endif
#endif

#if defined(USE_TDM)
#if __has_include(<hip/amd_detail/amd_gfx1250_TDM.h>)
#define TDM_6ARG 1   // therock-10.0 headers => clang-23 6-arg builtin
#endif

typedef unsigned int u32x4 __attribute__((ext_vector_type(4)));
typedef int          i32x4 __attribute__((ext_vector_type(4)));
typedef int          i32x8 __attribute__((ext_vector_type(8)));

__device__ __forceinline__ unsigned int lds_off(const void* p) {
  // Generic LDS addresses carry the byte offset in the low 32 bits
  // (flat aperture).
  return (unsigned int)(unsigned long long)p;
}

// 2-D tile load: tile (tileD1 rows x tileD0 elems) of 4-byte data from a
// row-major tensor with row stride strideD0 (elements). OOB reads (beyond
// tensorD0 / tensorD1 measured from the tile origin) return zero.
// LDS destination gets +1 DWORD padding per (1<<(padCode+1)) DWORDs.
__device__ __forceinline__ void tdm_load_2d(unsigned int ldsAddr,
                                            const void* gptr,
                                            unsigned int tensorD0,
                                            unsigned int tensorD1,
                                            unsigned int tileD0,
                                            unsigned int tileD1,
                                            unsigned long long strideD0,
                                            unsigned int padCode)
{
  const unsigned long long ga = (unsigned long long)gptr;
  u32x4 g0;
  g0[0] = 1u;                                              // count=1, user D#
  g0[1] = ldsAddr;                                         // lds_addr
  g0[2] = (unsigned int)(ga & 0xFFFFFFFFu);                // global_addr lo
  g0[3] = (unsigned int)((ga >> 32) & 0x01FFFFFFu)         // global_addr hi
          | (2u << 30);                                    // type=2 (image)
  i32x8 g1;
  unsigned int w0 = (2u << 16)                             // data_size = 4B
                  | (1u << 20)                             // pad_enable
                  | (padCode << 22);                       // pad_interval
                                                           // pad_amount=0 (1 DW)
  g1[0] = (int)w0;
  g1[1] = (int)((tensorD0 & 0xFFFFu) << 16);               // tdim0[15:0]
  g1[2] = (int)((tensorD0 >> 16) | ((tensorD1 & 0xFFFFu) << 16));
  g1[3] = (int)((tensorD1 >> 16) | (tileD0 << 16));
  g1[4] = (int)(tileD1 & 0xFFFFu);                         // tile_dim2 = 0
  g1[5] = (int)(unsigned int)(strideD0 & 0xFFFFFFFFu);
  g1[6] = (int)(unsigned int)((strideD0 >> 32) & 0xFFFFu); // dim1_stride = 0
  g1[7] = 0;
  const i32x4 z4 = {0, 0, 0, 0};
#if defined(TDM_6ARG)
  const i32x8 z8 = {0, 0, 0, 0, 0, 0, 0, 0};
  __builtin_amdgcn_tensor_load_to_lds(g0, g1, z4, z4, z8, 0);
#else
  __builtin_amdgcn_tensor_load_to_lds(g0, g1, z4, z4, 0);
#endif
}
#endif  // USE_TDM

// ---------------------------------------------------------------------------
// Generic WMMA GEMM:  C[b, cOff+m, n] = act( A[b, aOff+m, :K] @ W[:K, n] + bias[n] )
// W row-major (K x 128). One wave -> one 16x16 output tile; double-buffered
// TDM staging of the A (16x32) and W (32x16) f32 tiles.
// act: 0 = none, 1 = relu, 2 = tanh
// ---------------------------------------------------------------------------
__global__ __launch_bounds__(32)
void gemm_wmma(const float* __restrict__ A, long aBS, int aOff, int K,
               const float* __restrict__ W, const float* __restrict__ bias,
               float* __restrict__ C, long cBS, int cOff, int act)
{
  __shared__ float As32[2][16][33];   // 32 DW rows + 1 DW TDM pad
  __shared__ float Ws32[2][32][17];   // 16 DW rows + 1 DW TDM pad

  const int lane = threadIdx.x;
  const int hi   = lane >> 4;
  const int lo   = lane & 15;
  const int m0   = blockIdx.x * 16;
  const int n0   = blockIdx.y * 16;
  const int b    = blockIdx.z;

  const float* Ab = A + (size_t)b * aBS + (size_t)(aOff + m0) * K;
  v8f acc = {};
  const int nCh = (K + 31) / 32;

#if defined(USE_TDM)
  auto issue = [&](int c, int buf) {
    const int k0 = c * 32;
    // A tile: 16 rows x 32 k. tensorD0 = remaining elems => OOB k reads 0.
    tdm_load_2d(lds_off(&As32[buf][0][0]), Ab + k0,
                (unsigned)(K - k0), 16u, 32u, 16u,
                (unsigned long long)K, 4u /*pad every 32 DW*/);
    // W tile: 32 k-rows x 16 n. tensorD1 = remaining rows => OOB rows read 0.
    tdm_load_2d(lds_off(&Ws32[buf][0][0]), W + (size_t)k0 * kH + n0,
                (unsigned)(kH - n0), (unsigned)(K - k0), 16u, 32u,
                (unsigned long long)kH, 3u /*pad every 16 DW*/);
  };
  issue(0, 0);
  int cur = 0;
  for (int c = 0; c < nCh; ++c) {
    const bool hasNext = (c + 1 < nCh);
    if (hasNext) issue(c + 1, cur ^ 1);
    if (hasNext) __builtin_amdgcn_s_wait_tensorcnt(2);  // current 2 done
    else         __builtin_amdgcn_s_wait_tensorcnt(0);
    v16h af, bf;
    #pragma unroll
    for (int e = 0; e < 16; ++e) {
      const int ka = (hi ? 16 : 0) + e;
      af[e] = (_Float16)As32[cur][lo][ka];
      const int kb = (hi ? 16 : 0) + e;
      bf[e] = (_Float16)Ws32[cur][kb][lo];
    }
    acc = __builtin_amdgcn_wmma_f32_16x16x32_f16(false, af, false, bf,
                                                 (short)0, acc, false, false);
    cur ^= 1;
  }
#else
  for (int c = 0; c < nCh; ++c) {
    const int k0 = c * 32;
    #pragma unroll
    for (int t = 0; t < 16; ++t) {   // A: lanes cover k (coalesced)
      const int kk = k0 + lane;
      const int kc = (kk < K) ? kk : (K - 1);
      float v = Ab[(size_t)t * K + kc];
      As32[0][t][lane] = (kk < K) ? v : 0.0f;
    }
    #pragma unroll
    for (int e = 0; e < 16; ++e) {   // W: 32x16
      const int kr = e * 2 + hi;
      const int kk = k0 + kr;
      const int kc = (kk < K) ? kk : (K - 1);
      float v = W[(size_t)kc * kH + n0 + lo];
      Ws32[0][kr][lo] = (kk < K) ? v : 0.0f;
    }
    v16h af, bf;
    #pragma unroll
    for (int e = 0; e < 16; ++e) {
      const int ka = (hi ? 16 : 0) + e;
      af[e] = (_Float16)As32[0][lo][ka];
      const int kb = (hi ? 16 : 0) + e;
      bf[e] = (_Float16)Ws32[0][kb][lo];
    }
    acc = __builtin_amdgcn_wmma_f32_16x16x32_f16(false, af, false, bf,
                                                 (short)0, acc, false, false);
  }
#endif

  #pragma unroll
  for (int r = 0; r < 8; ++r) {
    const int mrow = r + (hi ? 8 : 0);
    const int col  = n0 + lo;
    float v = acc[r];
    if (bias) v += bias[col];
    if (act == 1)      v = fmaxf(v, 0.0f);
    else if (act == 2) v = tanhf(v);
    C[(size_t)b * cBS + (size_t)(cOff + m0 + mrow) * kH + col] = v;
  }
}

// ---------------------------------------------------------------------------
// src/dst per-head projections: src[b,n,h] = Wh[b,n,h,:] . a_src[h,:]
// ---------------------------------------------------------------------------
__global__ void srcdst_kernel(const float* __restrict__ Wh, int rowOff, int nSub,
                              const float* __restrict__ aSrc,
                              const float* __restrict__ aDst,
                              float* __restrict__ srcO, float* __restrict__ dstO)
{
  const int idx   = blockIdx.x * blockDim.x + threadIdx.x;
  const int total = kB * nSub * kHEADS;
  if (idx >= total) return;
  const int hh = idx % kHEADS;
  const int n  = (idx / kHEADS) % nSub;
  const int b  = idx / (kHEADS * nSub);
  const float* w = Wh + ((size_t)b * kN + rowOff + n) * kH + hh * kDIM;
  float s = 0.0f, d = 0.0f;
  #pragma unroll
  for (int t = 0; t < kDIM; ++t) {
    const float x = w[t];
    s += x * aSrc[hh * kDIM + t];
    d += x * aDst[hh * kDIM + t];
  }
  const size_t o = ((size_t)b * kN + rowOff + n) * kHEADS + hh;
  srcO[o] = s;
  dstO[o] = d;
}

// ---------------------------------------------------------------------------
// Fused GAT attention (flash-style online softmax + WMMA accumulate).
// Block = 128 threads = 4 waves; wave w handles head w. All LDS sharing is
// intra-wave (per-head tiles), so no block barriers are required.
// act: 1 = elu(out)   2 = relu(out + resid)
// ---------------------------------------------------------------------------
__global__ __launch_bounds__(128)
void gat_attn(const float* __restrict__ Wh,
              const float* __restrict__ srcB, const float* __restrict__ dstB,
              const float* __restrict__ adj, int rowOff, int nSub,
              const float* __restrict__ bias, const float* __restrict__ resid,
              int act, float* __restrict__ outP)
{
  __shared__ _Float16 pT[kHEADS][16][32];        // probability tiles
  __shared__ float    whS[kHEADS][2][32][33];    // Wh chunks (TDM padded)

  const int head = __builtin_amdgcn_readfirstlane(threadIdx.x >> 5);
  const int lane = threadIdx.x & 31;
  const int hi   = lane >> 4;
  const int lo   = lane & 15;
  const int i0   = blockIdx.x * 16;
  const int b    = blockIdx.y;

  const float* adjB = adj + (size_t)b * kN * kN;

  float srcv[16];
  #pragma unroll
  for (int i = 0; i < 16; ++i)
    srcv[i] = srcB[((size_t)b * kN + rowOff + i0 + i) * kHEADS + head];

  float m[16], s[16];
  #pragma unroll
  for (int i = 0; i < 16; ++i) { m[i] = -3.0e38f; s[i] = 0.0f; }
  v8f acc0 = {}, acc1 = {};

#if defined(USE_TDM)
  auto issueWh = [&](int j0v, int bufv) {
    // 32 rows x 32 f32, row stride kH, cols head*32..head*32+31
    tdm_load_2d(lds_off(&whS[head][bufv][0][0]),
                Wh + ((size_t)b * kN + rowOff + j0v) * kH + head * kDIM,
                (unsigned)(kH - head * kDIM), 32u, 32u, 32u,
                (unsigned long long)kH, 4u /*pad every 32 DW*/);
  };
  issueWh(0, 0);
#endif
  int cur = 0;

  for (int j0 = 0; j0 < nSub; j0 += 32) {
    const bool hasNext = (j0 + 32 < nSub);
#if defined(USE_TDM)
    if (hasNext) issueWh(j0 + 32, cur ^ 1);
#else
    {  // manual stage (single buffer)
      const float* wr =
          Wh + ((size_t)b * kN + rowOff + j0 + lane) * kH + head * kDIM;
      #pragma unroll
      for (int q = 0; q < 8; ++q) {
        const float4 v = *(const float4*)(wr + q * 4);
        whS[head][0][lane][q * 4 + 0] = v.x;
        whS[head][0][lane][q * 4 + 1] = v.y;
        whS[head][0][lane][q * 4 + 2] = v.z;
        whS[head][0][lane][q * 4 + 3] = v.w;
      }
    }
#endif
    const float dstv = dstB[((size_t)b * kN + rowOff + j0 + lane) * kHEADS + head];

    // Online softmax over this 32-wide j-chunk (overlaps the TDM transfer).
    float f[16];
    #pragma unroll
    for (int i = 0; i < 16; ++i) {
      float sc = srcv[i] + dstv;
      sc = (sc > 0.0f) ? sc : (kALPHA * sc);                 // leaky relu
      const size_t arow = (size_t)(rowOff + i0 + i) * kN + rowOff + j0 + lane;
      const float av = adjB[arow];
      if (hasNext) __builtin_prefetch(&adjB[arow + 32], 0, 0);
      sc = (av > 0.0f) ? sc : kNEG;                          // mask
      float cm = sc;
      #pragma unroll
      for (int off = 16; off > 0; off >>= 1) cm = fmaxf(cm, __shfl_xor(cm, off));
      const float mn = fmaxf(m[i], cm);
      f[i] = __expf(m[i] - mn);
      const float p = __expf(sc - mn);
      float ps = p;
      #pragma unroll
      for (int off = 16; off > 0; off >>= 1) ps += __shfl_xor(ps, off);
      s[i] = s[i] * f[i] + ps;
      m[i] = mn;
      pT[head][i][lane] = (_Float16)p;
    }
    // Rescale running accumulators to the new max.
    #pragma unroll
    for (int r = 0; r < 8; ++r) {
      const float fr = f[r + (hi ? 8 : 0)];
      acc0[r] *= fr; acc1[r] *= fr;
    }

#if defined(USE_TDM)
    if (hasNext) __builtin_amdgcn_s_wait_tensorcnt(1);  // current chunk done
    else         __builtin_amdgcn_s_wait_tensorcnt(0);
#endif
    v16h af, b0, b1;
    #pragma unroll
    for (int e = 0; e < 16; ++e) {
      const int ka = (hi ? 16 : 0) + e;
      af[e] = pT[head][lo][ka];
      const int kb = (hi ? 16 : 0) + e;
      b0[e] = (_Float16)whS[head][cur][kb][lo];
      b1[e] = (_Float16)whS[head][cur][kb][16 + lo];
    }
    acc0 = __builtin_amdgcn_wmma_f32_16x16x32_f16(false, af, false, b0,
                                                  (short)0, acc0, false, false);
    acc1 = __builtin_amdgcn_wmma_f32_16x16x32_f16(false, af, false, b1,
                                                  (short)0, acc1, false, false);
#if defined(USE_TDM)
    cur ^= 1;
#endif
  }

  #pragma unroll
  for (int r = 0; r < 8; ++r) {
    const int row = r + (hi ? 8 : 0);
    const float rs = (s[row] > 0.0f) ? (1.0f / s[row]) : 0.0f;
    const size_t base = ((size_t)b * kN + rowOff + i0 + row) * kH + head * kDIM;
    float v0 = acc0[r] * rs + bias[head * kDIM + lo];
    float v1 = acc1[r] * rs + bias[head * kDIM + 16 + lo];
    if (act == 1) {            // elu
      v0 = (v0 > 0.0f) ? v0 : (__expf(v0) - 1.0f);
      v1 = (v1 > 0.0f) ? v1 : (__expf(v1) - 1.0f);
    } else {                   // relu(out + residual)
      v0 = fmaxf(v0 + resid[base + lo], 0.0f);
      v1 = fmaxf(v1 + resid[base + 16 + lo], 0.0f);
    }
    outP[base + lo]      = v0;
    outP[base + 16 + lo] = v1;
  }
}

// ---------------------------------------------------------------------------
// Readout: s_n = T[n,:].v + mask_n ; a = softmax(s) ; r = sum_n a_n h[n,:]
// ---------------------------------------------------------------------------
__global__ __launch_bounds__(256)
void readout_kernel(const float* __restrict__ T, const float* __restrict__ hF,
                    int rowOff, const float* __restrict__ v,
                    const int* __restrict__ lengths,
                    float* __restrict__ outP, int outBS)
{
  __shared__ float sarr[kL];
  __shared__ float red[256];
  const int tid = threadIdx.x;
  const int b   = blockIdx.x;
  const int len = lengths[b];

  for (int n = tid; n < kL; n += 256) {
    const float* tr = T + ((size_t)b * kN + rowOff + n) * kH;
    float acc = 0.0f;
    for (int h2 = 0; h2 < kH; h2 += 4) {
      const float4 tv = *(const float4*)(tr + h2);
      const float4 vv = *(const float4*)(v + h2);
      acc += tv.x * vv.x + tv.y * vv.y + tv.z * vv.z + tv.w * vv.w;
    }
    sarr[n] = acc + ((n < len) ? 0.0f : kNEG);
  }
  __syncthreads();

  float lm = -3.0e38f;
  for (int n = tid; n < kL; n += 256) lm = fmaxf(lm, sarr[n]);
  red[tid] = lm; __syncthreads();
  for (int st = 128; st > 0; st >>= 1) {
    if (tid < st) red[tid] = fmaxf(red[tid], red[tid + st]);
    __syncthreads();
  }
  const float M = red[0]; __syncthreads();

  float ls = 0.0f;
  for (int n = tid; n < kL; n += 256) {
    const float e = __expf(sarr[n] - M);
    sarr[n] = e;
    ls += e;
  }
  red[tid] = ls; __syncthreads();
  for (int st = 128; st > 0; st >>= 1) {
    if (tid < st) red[tid] += red[tid + st];
    __syncthreads();
  }
  const float S = red[0]; __syncthreads();

  if (tid < kH) {
    float acc = 0.0f;
    for (int n = 0; n < kL; ++n)
      acc += sarr[n] * hF[((size_t)b * kN + rowOff + n) * kH + tid];
    outP[(size_t)b * outBS + tid] = acc / S;
  }
}

// ---------------------------------------------------------------------------
// L2 row-normalize. If s1 != null: rows interleave stack([s0, s1], axis=1).
// ---------------------------------------------------------------------------
__global__ __launch_bounds__(128)
void norm_rows(const float* __restrict__ s0, const float* __restrict__ s1,
               float* __restrict__ dst, int rowlen)
{
  __shared__ float red[128];
  const int r = blockIdx.x;
  const float* src;
  if (s1) {
    const int b = r >> 1, k = r & 1;
    src = (k ? s1 : s0) + (size_t)b * rowlen;
  } else {
    src = s0 + (size_t)r * rowlen;
  }
  float ss = 0.0f;
  for (int i = threadIdx.x; i < rowlen; i += 128) { const float x = src[i]; ss += x * x; }
  red[threadIdx.x] = ss; __syncthreads();
  for (int st = 64; st > 0; st >>= 1) {
    if (threadIdx.x < st) red[threadIdx.x] += red[threadIdx.x + st];
    __syncthreads();
  }
  const float inv = 1.0f / sqrtf(red[0]);
  for (int i = threadIdx.x; i < rowlen; i += 128)
    dst[(size_t)r * rowlen + i] = src[i] * inv;
}

// ---------------------------------------------------------------------------
// Final FC: out[b] = rm[b,:] . Wf + bf
// ---------------------------------------------------------------------------
__global__ __launch_bounds__(128)
void fc_kernel(const float* __restrict__ rm, const float* __restrict__ Wf,
               const float* __restrict__ bf, float* __restrict__ outP)
{
  __shared__ float red[128];
  const int b = blockIdx.x;
  float acc = 0.0f;
  for (int i = threadIdx.x; i < 3 * kH; i += 128)
    acc += rm[(size_t)b * 3 * kH + i] * Wf[i];
  red[threadIdx.x] = acc; __syncthreads();
  for (int st = 64; st > 0; st >>= 1) {
    if (threadIdx.x < st) red[threadIdx.x] += red[threadIdx.x + st];
    __syncthreads();
  }
  if (threadIdx.x == 0) outP[b] = red[0] + bf[0];
}

// ---------------------------------------------------------------------------
// Host-side orchestration (input order = setup_inputs() insertion order).
// ---------------------------------------------------------------------------
struct GLayer { const float *W, *asrc, *adst, *b; };
struct Agg    { const float *W, *b, *v; };

extern "C" void kernel_launch(void* const* d_in, const int* in_sizes, int n_in,
                              void* d_out, int out_size, void* d_ws, size_t ws_size,
                              hipStream_t stream)
{
  (void)in_sizes; (void)n_in; (void)out_size; (void)ws_size;

  const float* text    = (const float*)d_in[0];
  const float* video   = (const float*)d_in[1];
  const float* audio   = (const float*)d_in[2];
  const int*   lengths = (const int*)d_in[3];
  const float* adjA    = (const float*)d_in[4];
  const float* adjG    = (const float*)d_in[5];

  const float* proj_tW = (const float*)d_in[6];  const float* proj_tb = (const float*)d_in[7];
  const float* proj_vW = (const float*)d_in[8];  const float* proj_vb = (const float*)d_in[9];
  const float* proj_aW = (const float*)d_in[10]; const float* proj_ab = (const float*)d_in[11];

  GLayer gat_t[2], gat_v[2], gat_a[2], gat_m[2];
  auto ldG = [&](GLayer* g, int base) {
    for (int l = 0; l < 2; ++l) {
      g[l].W    = (const float*)d_in[base + 4 * l + 0];
      g[l].asrc = (const float*)d_in[base + 4 * l + 1];
      g[l].adst = (const float*)d_in[base + 4 * l + 2];
      g[l].b    = (const float*)d_in[base + 4 * l + 3];
    }
  };
  ldG(gat_t, 12); ldG(gat_v, 20); ldG(gat_a, 28); ldG(gat_m, 36);

  Agg ro[3];
  for (int i = 0; i < 3; ++i) {
    ro[i].W = (const float*)d_in[44 + 3 * i + 0];
    ro[i].b = (const float*)d_in[44 + 3 * i + 1];
    ro[i].v = (const float*)d_in[44 + 3 * i + 2];
  }
  const float* fcW = (const float*)d_in[53];
  const float* fcb = (const float*)d_in[54];

  // ---- workspace carve (floats), ~10 MB ----
  const size_t NH = (size_t)kN * kH;
  float* ws   = (float*)d_ws;
  float* hs   = ws; ws += (size_t)kB * NH;
  float* htmp = ws; ws += (size_t)kB * NH;
  float* gbuf = ws; ws += (size_t)kB * NH;
  float* WhB  = ws; ws += (size_t)kB * NH;
  float* srcB = ws; ws += (size_t)kB * kN * kHEADS;
  float* dstB = ws; ws += (size_t)kB * kN * kHEADS;
  float* rmR[2]; float* rtR[2]; float* rvR[2]; float* raR[2];
  for (int br = 0; br < 2; ++br) {
    rmR[br] = ws; ws += (size_t)kB * 3 * kH;
    rtR[br] = ws; ws += (size_t)kB * kH;
    rvR[br] = ws; ws += (size_t)kB * kH;
    raR[br] = ws; ws += (size_t)kB * kH;
  }

  auto gemm = [&](const float* A, long aBS, int aOff, int K, int Mrows,
                  const float* W, const float* bias, float* C, int cOff, int act) {
    dim3 g(Mrows / 16, kH / 16, kB);
    gemm_wmma<<<g, dim3(32), 0, stream>>>(A, aBS, aOff, K, W, bias,
                                          C, (long)NH, cOff, act);
  };
  auto srcdst = [&](int off, int nSub, const GLayer& gl) {
    const int total = kB * nSub * kHEADS;
    srcdst_kernel<<<(total + 255) / 256, 256, 0, stream>>>(
        WhB, off, nSub, gl.asrc, gl.adst, srcB, dstB);
  };
  auto attn = [&](const float* adj, int off, int nSub, const float* bias,
                  const float* resid, int act, float* outP) {
    gat_attn<<<dim3(nSub / 16, kB), dim3(128), 0, stream>>>(
        WhB, srcB, dstB, adj, off, nSub, bias, resid, act, outP);
  };
  auto readout = [&](const float* T, const float* hF, int off, const float* v,
                     float* outP, int obs) {
    readout_kernel<<<dim3(kB), dim3(256), 0, stream>>>(T, hF, off, v, lengths,
                                                       outP, obs);
  };

  // ---- projections -> hs (B, N=3L, H) ----
  gemm(text,  (long)kL * kDT,  0, kDT,  kL, proj_tW, proj_tb, hs, 0,      1);
  gemm(video, (long)kL * kDV2, 0, kDV2, kL, proj_vW, proj_vb, hs, kL,     1);
  gemm(audio, (long)kL * kDA2, 0, kDA2, kL, proj_aW, proj_ab, hs, 2 * kL, 1);

  const GLayer* gmods[3] = { gat_t, gat_v, gat_a };

  auto runBranch = [&](const float* adj, float* rm, float* rt, float* rv, float* ra) {
    // gat_m over the full multimodal graph
    gemm(hs,   (long)NH, 0, kH, kN, gat_m[0].W, nullptr, WhB, 0, 0);
    srcdst(0, kN, gat_m[0]);
    attn(adj, 0, kN, gat_m[0].b, nullptr, 1, htmp);
    gemm(htmp, (long)NH, 0, kH, kN, gat_m[1].W, nullptr, WhB, 0, 0);
    srcdst(0, kN, gat_m[1]);
    attn(adj, 0, kN, gat_m[1].b, hs, 2, gbuf);     // g = relu(out + hs)
    // rm = concat of segment readouts of g
    for (int seg = 0; seg < 3; ++seg) {
      gemm(gbuf, (long)NH, seg * kL, kH, kL, ro[seg].W, ro[seg].b, WhB, seg * kL, 2);
      readout(WhB, gbuf, seg * kL, ro[seg].v, rm + seg * kH, 3 * kH);
    }
    // per-modality GATs on diagonal adjacency blocks
    for (int mo = 0; mo < 3; ++mo) {
      const int off = mo * kL;
      const GLayer* gp = gmods[mo];
      gemm(hs,   (long)NH, off, kH, kL, gp[0].W, nullptr, WhB, off, 0);
      srcdst(off, kL, gp[0]);
      attn(adj, off, kL, gp[0].b, nullptr, 1, htmp);
      gemm(htmp, (long)NH, off, kH, kL, gp[1].W, nullptr, WhB, off, 0);
      srcdst(off, kL, gp[1]);
      attn(adj, off, kL, gp[1].b, hs, 2, gbuf);    // relu(g + hs_mod)
      float* outr = (mo == 0) ? rt : ((mo == 1) ? rv : ra);
      gemm(gbuf, (long)NH, off, kH, kL, ro[mo].W, ro[mo].b, WhB, off, 2);
      readout(WhB, gbuf, off, ro[mo].v, outr, kH);
    }
  };

  runBranch(adjA, rmR[0], rtR[0], rvR[0], raR[0]);  // branch(adj_matrix)
  runBranch(adjG, rmR[1], rtR[1], rvR[1], raR[1]);  // branch(adj_matrix_aug)

  // ---- final outputs ----
  float* out = (float*)d_out;
  fc_kernel<<<dim3(kB), dim3(128), 0, stream>>>(rmR[0], fcW, fcb, out + 0);
  norm_rows<<<dim3(kB), dim3(128), 0, stream>>>(rmR[0], nullptr, out + 4,    3 * kH);
  norm_rows<<<dim3(kB), dim3(128), 0, stream>>>(rtR[0], nullptr, out + 1540, kH);
  norm_rows<<<dim3(kB), dim3(128), 0, stream>>>(rvR[0], nullptr, out + 2052, kH);
  norm_rows<<<dim3(kB), dim3(128), 0, stream>>>(raR[0], nullptr, out + 2564, kH);
  // stacked: index 0 = aug branch, index 1 = original branch
  norm_rows<<<dim3(2 * kB), dim3(128), 0, stream>>>(rmR[1], rmR[0], out + 3076, 3 * kH);
  norm_rows<<<dim3(2 * kB), dim3(128), 0, stream>>>(rtR[1], rtR[0], out + 6148, kH);
  norm_rows<<<dim3(2 * kB), dim3(128), 0, stream>>>(rvR[1], rvR[0], out + 7172, kH);
  norm_rows<<<dim3(2 * kB), dim3(128), 0, stream>>>(raR[1], raR[0], out + 8196, kH);
}
